// UnderwaterFasterRCNN_78658031059360
// MI455X (gfx1250) — compile-verified, matrix operand order checked by reading, not run on hardware
//
#include <hip/hip_runtime.h>
#include <hip/hip_bf16.h>

// ---------------------------------------------------------------------------
// CDNA5 wave32 WMMA types (plain ext-vector types: trivial, union-legal).
// ---------------------------------------------------------------------------
typedef __bf16 bf16_t;
typedef __attribute__((ext_vector_type(16))) __bf16        v16bf;
typedef __attribute__((ext_vector_type(8)))  float         v8f;
typedef __attribute__((ext_vector_type(4)))  unsigned int  v4u;

union BF16Frag {
    bf16_t h[16];
    v4u    q[2];
    v16bf  v;
};

// ---------------------------------------------------------------------------
// A-fragment loaders for v_wmma_f32_16x16x32_bf16:
// lane L holds row M = L&15, half = L>>4; 16 elements are the two contiguous
// k-chunks [8*half, 8*half+8) and [16+8*half, 16+8*half+8).
// bf16 source: two 16-byte loads, no ALU. fp32 source: hardware cvt casts.
// ---------------------------------------------------------------------------
__device__ __forceinline__ void load_afrag(BF16Frag& a, const bf16_t* p) {
    a.q[0] = *(const v4u*)p;
    a.q[1] = *(const v4u*)(p + 16);
}
__device__ __forceinline__ void load_afrag(BF16Frag& a, const float* p) {
#pragma unroll
    for (int j = 0; j < 8; ++j) {
        a.h[j]     = (bf16_t)p[j];
        a.h[j + 8] = (bf16_t)p[j + 16];
    }
}

// ---------------------------------------------------------------------------
// Weight packing: W[t][k][n] (row-major over n) -> per-lane B-fragment layout
//   packed[(((t*NT + nt)*KS + ks)*32 + lane)*16 + j],
//   n = nt*16 + (lane&15), half = lane>>4,
//   k = ks*32 + (j<8 ? 8*half + j : 16 + 8*half + (j-8)).
// Columns >= Nsrc zero-padded (3/8/12-wide heads run at N=16).
// ---------------------------------------------------------------------------
__global__ __launch_bounds__(256)
void pack_weights_kernel(const float* __restrict__ src, bf16_t* __restrict__ dst,
                         int T, int K, int Nsrc, int Npad) {
    const int KS = K >> 5, NT = Npad >> 4;
    const size_t total = (size_t)T * NT * KS * 512;
    size_t idx = (size_t)blockIdx.x * blockDim.x + threadIdx.x;
    if (idx >= total) return;
    int j    = (int)(idx & 15);
    int lane = (int)((idx >> 4) & 31);
    size_t rest = idx >> 9;
    int ks = (int)(rest % KS); rest /= KS;
    int nt = (int)(rest % NT);
    int t  = (int)(rest / NT);
    int half = lane >> 4;
    int k = ks * 32 + ((j < 8) ? (half * 8 + j) : (16 + half * 8 + (j - 8)));
    int n = nt * 16 + (lane & 15);
    float v = (n < Nsrc) ? src[((size_t)t * K + k) * Nsrc + n] : 0.0f;
    dst[idx] = (bf16_t)v;
}

// ---------------------------------------------------------------------------
// D-tile store (C/D layout: VGPR v -> row v + 8*(lane>=16), col = lane&15).
// act: 0=none, 1=relu, 2=sigmoid. OT selects fp32 (final outs) or bf16 (ws).
// ---------------------------------------------------------------------------
template<typename OT, int JT>
__device__ __forceinline__
void store_tile(const v8f* acc, OT* __restrict__ C, int ldc,
                int mtile, int ntbase, int M, int Nreal,
                const float* __restrict__ bias, int act, int lane) {
    const int chalf = (lane >> 4) * 8;
    const int col0  = lane & 15;
#pragma unroll
    for (int j = 0; j < JT; ++j) {
        int col = (ntbase + j) * 16 + col0;
        if (col >= Nreal) continue;
        float bv = bias[col];
#pragma unroll
        for (int v = 0; v < 8; ++v) {
            int r = mtile * 16 + chalf + v;
            if (r >= M) continue;
            float x = acc[j][v] + bv;
            if (act == 1)      x = fmaxf(x, 0.0f);
            else if (act == 2) x = 1.0f / (1.0f + expf(-x));
            C[(size_t)r * ldc + col] = (OT)x;
        }
    }
}

// ---------------------------------------------------------------------------
// GEMM: C[M,Nreal] = act(A[M,K] * Wp + bias). One wave -> 16 x (16*JT) tile;
// A fragment reused across JT branch-free WMMAs. Rows clamped (D row m only
// depends on A row m), so no per-element edge selects. 4 waves/block on
// consecutive M-tiles. grid.y * JT * 16 == Npad of the packed weights.
// ---------------------------------------------------------------------------
template<typename AT, typename OT, int JT>
__global__ __launch_bounds__(128)
void wmma_gemm_kernel(const AT* __restrict__ A, int lda,
                      const bf16_t* __restrict__ Bp,
                      const float* __restrict__ bias,
                      OT* __restrict__ C, int ldc,
                      int M, int K, int Nreal, int act) {
    const int lane   = threadIdx.x & 31;
    const int wave   = threadIdx.x >> 5;
    const int mtile  = blockIdx.x * 4 + wave;
    const int ntbase = blockIdx.y * JT;
    const int KS = K >> 5;
    if (mtile * 16 >= M) return;                 // uniform per wave (no barriers)

    int row = mtile * 16 + (lane & 15);
    row = row < M ? row : M - 1;                 // clamp: safe, row-local
    const int khalf = (lane >> 4) * 8;
    const AT* arow = A + (size_t)row * lda + khalf;

    v8f acc[JT];
#pragma unroll
    for (int j = 0; j < JT; ++j)
        acc[j] = v8f{0.f,0.f,0.f,0.f,0.f,0.f,0.f,0.f};

    for (int ks = 0; ks < KS; ++ks) {
        const AT* ap = arow + (size_t)ks * 32;
        if (ks + 1 < KS) __builtin_prefetch(ap + 32, 0, 0);   // global_prefetch_b8
        BF16Frag a;
        load_afrag(a, ap);
#pragma unroll
        for (int j = 0; j < JT; ++j) {
            BF16Frag b;
            const v4u* bp = (const v4u*)(Bp + (((size_t)(ntbase + j) * KS + ks) * 32 + lane) * 16);
            b.q[0] = bp[0]; b.q[1] = bp[1];
            acc[j] = __builtin_amdgcn_wmma_f32_16x16x32_bf16(
                false, a.v, false, b.v, (short)0, acc[j], false, false);
        }
    }
    store_tile<OT, JT>(acc, C, ldc, mtile, ntbase, M, Nreal, bias, act, lane);
}

// ---------------------------------------------------------------------------
// Implicit-GEMM 3x3 conv, NHWC bf16 in / bf16 out, SAME (pad=1), stride 1|2.
// Per (tap, k-step) the block's shared B panel (JT fragments, identical for
// all 4 waves) is staged into LDS with gfx1250 async copies
// (global_load_async_to_lds_b128; INST_OFFSET applies to both LDS and global
// addresses), waited with s_wait_asynccnt, then consumed via ds_load_b128.
// All waves stay alive (clamped rows) so workgroup barriers are safe.
// ---------------------------------------------------------------------------
template<int JT>
__global__ __launch_bounds__(128)
void wmma_conv3x3_kernel(const bf16_t* __restrict__ X,
                         int Nb, int H, int W, int Cin,
                         const bf16_t* __restrict__ Wp,
                         const float* __restrict__ bias,
                         bf16_t* __restrict__ Y,
                         int Ho, int Wo, int Cout, int stride, int act) {
    __shared__ bf16_t sB[JT * 32 * 16];
    const int tid    = threadIdx.x;
    const int lane   = tid & 31;
    const int wave   = tid >> 5;
    const int mtile  = blockIdx.x * 4 + wave;
    const int ntbase = blockIdx.y * JT;
    const int KS     = Cin >> 5;
    const int NTtot  = Cout >> 4;
    const int M      = Nb * Ho * Wo;

    const int  m0 = mtile * 16 + (lane & 15);
    const bool mv = m0 < M;
    const int  mc = mv ? m0 : M - 1;
    const int  wo = mc % Wo;
    const int  t1 = mc / Wo;
    const int  ho = t1 % Ho;
    const int  nb = t1 / Ho;
    const int  khalf = (lane >> 4) * 8;
    const v4u  vzero = {0u, 0u, 0u, 0u};

    v8f acc[JT];
#pragma unroll
    for (int j = 0; j < JT; ++j)
        acc[j] = v8f{0.f,0.f,0.f,0.f,0.f,0.f,0.f,0.f};

    for (int tap = 0; tap < 9; ++tap) {
        const int dy = tap / 3, dx = tap % 3;
        const int h = ho * stride + dy - 1;
        const int w = wo * stride + dx - 1;
        const bool inb = mv && (h >= 0) && (h < H) && (w >= 0) && (w < W);
        const int hc = h < 0 ? 0 : (h >= H ? H - 1 : h);
        const int wc = w < 0 ? 0 : (w >= W ? W - 1 : w);
        const bf16_t* xp = X + (((size_t)nb * H + hc) * W + wc) * Cin + khalf;

        for (int ks = 0; ks < KS; ++ks) {
            // ---- async-stage B panel for this (tap, ks): JT * 1KB ----
            for (int e = tid; e < JT * 32; e += 128) {
                const int jj = e >> 5, l = e & 31;
                const bf16_t* g = Wp +
                    ((((size_t)tap * NTtot + ntbase + jj) * KS + ks) * 32 + l) * 16;
                unsigned           ldsa = (unsigned)(size_t)&sB[(size_t)e * 16];
                unsigned long long ga   = (unsigned long long)(const void*)g;
                asm volatile("global_load_async_to_lds_b128 %0, %1, off"
                             :: "v"(ldsa), "v"(ga) : "memory");
                asm volatile("global_load_async_to_lds_b128 %0, %1, off offset:16"
                             :: "v"(ldsa), "v"(ga) : "memory");
            }
            asm volatile("s_wait_asynccnt 0x0" ::: "memory");
            __syncthreads();

            // ---- A fragment: two b128 loads + panel-edge zero select ----
            BF16Frag a;
            const bf16_t* ap = xp + ks * 32;
            a.q[0] = *(const v4u*)ap;
            a.q[1] = *(const v4u*)(ap + 16);
            a.q[0] = inb ? a.q[0] : vzero;
            a.q[1] = inb ? a.q[1] : vzero;

#pragma unroll
            for (int j = 0; j < JT; ++j) {
                BF16Frag b;
                const v4u* q = (const v4u*)&sB[((j * 32) + lane) * 16];
                b.q[0] = q[0]; b.q[1] = q[1];
                acc[j] = __builtin_amdgcn_wmma_f32_16x16x32_bf16(
                    false, a.v, false, b.v, (short)0, acc[j], false, false);
            }
            __syncthreads();
        }
    }
    store_tile<bf16_t, JT>(acc, Y, Cout, mtile, ntbase, M, Cout, bias, act, lane);
}

// ---------------------------------------------------------------------------
// lower[n,h,w,c] += upper[n,h/2,w/2,c] on bf16 tensors (fused up2 + add).
// ---------------------------------------------------------------------------
__global__ __launch_bounds__(256)
void upadd_kernel(bf16_t* __restrict__ lower, const bf16_t* __restrict__ upper,
                  int Nb, int H, int W, int C) {
    size_t idx = (size_t)blockIdx.x * blockDim.x + threadIdx.x;
    size_t tot = (size_t)Nb * H * W * C;
    if (idx >= tot) return;
    int c = (int)(idx % C); size_t t = idx / C;
    int w = (int)(t % W);   t /= W;
    int h = (int)(t % H);
    int n = (int)(t / H);
    int Hu = H >> 1, Wu = W >> 1;
    float s = (float)lower[idx]
            + (float)upper[(((size_t)n * Hu + (h >> 1)) * Wu + (w >> 1)) * C + c];
    lower[idx] = (bf16_t)s;
}

// ---------------------------------------------------------------------------
// TF crop_and_resize (bilinear, extrapolation 0) on bf16 fm=[50,50,256],
// 7x7 grid + mean. One block per ROI, one thread per channel.
// ---------------------------------------------------------------------------
__global__ __launch_bounds__(256)
void roi_pool_kernel(const bf16_t* __restrict__ fm, const float* __restrict__ rois,
                     bf16_t* __restrict__ pooled) {
    const int roi = blockIdx.x;
    const int c   = threadIdx.x;
    const float inv50 = 1.0f / 50.0f;
    float y1 = rois[roi * 4 + 0] * inv50;
    float x1 = rois[roi * 4 + 1] * inv50;
    float y2 = rois[roi * 4 + 2] * inv50;
    float x2 = rois[roi * 4 + 3] * inv50;
    const float S = 49.0f, inv6 = 1.0f / 6.0f;
    float sum = 0.0f;
    for (int iy = 0; iy < 7; ++iy) {
        float yc = y1 * S + (float)iy * ((y2 - y1) * S * inv6);
        float yf = floorf(yc);
        float wy = yc - yf;
        int y0 = (int)yf; y0 = y0 < 0 ? 0 : (y0 > 49 ? 49 : y0);
        int y1i = y0 + 1 > 49 ? 49 : y0 + 1;
        bool vy = (yc >= 0.0f) && (yc <= 49.0f);
        for (int ix = 0; ix < 7; ++ix) {
            float xc = x1 * S + (float)ix * ((x2 - x1) * S * inv6);
            float xf = floorf(xc);
            float wx = xc - xf;
            int x0 = (int)xf; x0 = x0 < 0 ? 0 : (x0 > 49 ? 49 : x0);
            int x1b = x0 + 1 > 49 ? 49 : x0 + 1;
            bool vx = (xc >= 0.0f) && (xc <= 49.0f);
            if (vy && vx) {
                float a00 = (float)fm[((size_t)y0  * 50 + x0 ) * 256 + c];
                float a01 = (float)fm[((size_t)y0  * 50 + x1b) * 256 + c];
                float a10 = (float)fm[((size_t)y1i * 50 + x0 ) * 256 + c];
                float a11 = (float)fm[((size_t)y1i * 50 + x1b) * 256 + c];
                sum += (1.f - wy) * (1.f - wx) * a00 + (1.f - wy) * wx * a01
                     +  wy       * (1.f - wx) * a10 +  wy       * wx * a11;
            }
        }
    }
    pooled[(size_t)roi * 256 + c] = (bf16_t)(sum * (1.0f / 49.0f));
}

// ---------------------------------------------------------------------------
// Row-wise softmax over 8 classes (512 rows).
// ---------------------------------------------------------------------------
__global__ __launch_bounds__(256)
void softmax8_kernel(const float* __restrict__ logits, float* __restrict__ out) {
    int r = blockIdx.x * blockDim.x + threadIdx.x;
    if (r >= 512) return;
    const float* p = logits + (size_t)r * 8;
    float m = p[0];
#pragma unroll
    for (int i = 1; i < 8; ++i) m = fmaxf(m, p[i]);
    float e[8], s = 0.0f;
#pragma unroll
    for (int i = 0; i < 8; ++i) { e[i] = expf(p[i] - m); s += e[i]; }
    float inv = 1.0f / s;
#pragma unroll
    for (int i = 0; i < 8; ++i) out[(size_t)r * 8 + i] = e[i] * inv;
}

// ---------------------------------------------------------------------------
// Anchor generation (5 levels x 3 ratios), matching make_anchors().
// ---------------------------------------------------------------------------
__global__ __launch_bounds__(256)
void anchors_kernel(float* __restrict__ out) {
    int idx = blockIdx.x * blockDim.x + threadIdx.x;
    if (idx >= 159882) return;
    const int   wdim[5]  = {200, 100, 50, 25, 13};
    const int   start[5] = {0, 120000, 150000, 157500, 159375};
    const float strd[5]  = {4.f, 8.f, 16.f, 32.f, 64.f};
    const float scal[5]  = {32.f, 64.f, 128.f, 256.f, 512.f};
    int lev = 4;
#pragma unroll
    for (int l = 0; l < 4; ++l)
        if (idx < start[l + 1]) { lev = l; break; }
    int li   = idx - start[lev];
    int r    = li % 3;
    int cell = li / 3;
    int wd = wdim[lev];
    int x = cell % wd, y = cell / wd;
    float ratio = (r == 0) ? 0.5f : ((r == 1) ? 1.0f : 2.0f);
    float s  = scal[lev];
    float sq = sqrtf(ratio);
    float bx = 0.5f * s * sq;
    float by = 0.5f * s / sq;
    float sx = (float)x * strd[lev], sy = (float)y * strd[lev];
    float* o = out + (size_t)idx * 4;
    o[0] = sx - bx; o[1] = sy - by; o[2] = sx + bx; o[3] = sy + by;
}

// ---------------------------------------------------------------------------
// Host-side orchestration.
// ---------------------------------------------------------------------------
extern "C" void kernel_launch(void* const* d_in, const int* in_sizes, int n_in,
                              void* d_out, int out_size, void* d_ws, size_t ws_size,
                              hipStream_t stream) {
    (void)in_sizes; (void)n_in; (void)out_size; (void)ws_size;

    const float* feat2 = (const float*)d_in[0];
    const float* feat3 = (const float*)d_in[1];
    const float* feat4 = (const float*)d_in[2];
    const float* feat5 = (const float*)d_in[3];
    const float* rois  = (const float*)d_in[4];
    const float* lw2 = (const float*)d_in[5];  const float* lb2 = (const float*)d_in[6];
    const float* lw3 = (const float*)d_in[7];  const float* lb3 = (const float*)d_in[8];
    const float* lw4 = (const float*)d_in[9];  const float* lb4 = (const float*)d_in[10];
    const float* lw5 = (const float*)d_in[11]; const float* lb5 = (const float*)d_in[12];
    const float* fw  = (const float*)d_in[13]; const float* fb  = (const float*)d_in[14];
    const float* p6w = (const float*)d_in[15]; const float* p6b = (const float*)d_in[16];
    const float* rpnw = (const float*)d_in[17]; const float* rpnb = (const float*)d_in[18];
    const float* clsw = (const float*)d_in[19]; const float* clsb = (const float*)d_in[20];
    const float* regw = (const float*)d_in[21]; const float* regb = (const float*)d_in[22];
    const float* fc1w = (const float*)d_in[23]; const float* fc1b = (const float*)d_in[24];
    const float* fc2w = (const float*)d_in[25]; const float* fc2b = (const float*)d_in[26];
    const float* hclsw = (const float*)d_in[27]; const float* hclsb = (const float*)d_in[28];
    const float* hregw = (const float*)d_in[29]; const float* hregb = (const float*)d_in[30];
    float* out = (float*)d_out;

    // ---- workspace bump allocator ----
    char* wsb = (char*)d_ws;
    size_t cur = 0;
    auto alloc = [&](size_t bytes) -> void* {
        void* p = wsb + cur;
        cur += (bytes + 255) & ~(size_t)255;
        return p;
    };

    // packed bf16 weights (element counts = T * Npad * K)
    bf16_t* pw_l2  = (bf16_t*)alloc((size_t)256  * 256  * 2);
    bf16_t* pw_l3  = (bf16_t*)alloc((size_t)512  * 256  * 2);
    bf16_t* pw_l4  = (bf16_t*)alloc((size_t)1024 * 256  * 2);
    bf16_t* pw_l5  = (bf16_t*)alloc((size_t)2048 * 256  * 2);
    bf16_t* pw_f[4];
    for (int l = 0; l < 4; ++l)
        pw_f[l] = (bf16_t*)alloc((size_t)9 * 256 * 256 * 2);
    bf16_t* pw_p6  = (bf16_t*)alloc((size_t)9 * 256 * 256 * 2);
    bf16_t* pw_rpn = (bf16_t*)alloc((size_t)9 * 256 * 512 * 2);
    bf16_t* pw_cls = (bf16_t*)alloc((size_t)512 * 16 * 2);
    bf16_t* pw_reg = (bf16_t*)alloc((size_t)512 * 16 * 2);
    bf16_t* pw_fc1 = (bf16_t*)alloc((size_t)256 * 1024 * 2);
    bf16_t* pw_fc2 = (bf16_t*)alloc((size_t)1024 * 1024 * 2);
    bf16_t* pw_hcl = (bf16_t*)alloc((size_t)1024 * 16 * 2);
    bf16_t* pw_hrg = (bf16_t*)alloc((size_t)1024 * 32 * 2);

    // bf16 activations (halved HBM traffic vs fp32)
    bf16_t* l2 = (bf16_t*)alloc((size_t)2 * 200 * 200 * 256 * 2);
    bf16_t* l3 = (bf16_t*)alloc((size_t)2 * 100 * 100 * 256 * 2);
    bf16_t* l4 = (bf16_t*)alloc((size_t)2 * 50  * 50  * 256 * 2);
    bf16_t* l5 = (bf16_t*)alloc((size_t)2 * 25  * 25  * 256 * 2);
    bf16_t* p2 = (bf16_t*)alloc((size_t)2 * 200 * 200 * 256 * 2);
    bf16_t* p3 = (bf16_t*)alloc((size_t)2 * 100 * 100 * 256 * 2);
    bf16_t* p4 = (bf16_t*)alloc((size_t)2 * 50  * 50  * 256 * 2);
    bf16_t* p5 = (bf16_t*)alloc((size_t)2 * 25  * 25  * 256 * 2);
    bf16_t* p6 = (bf16_t*)alloc((size_t)2 * 13  * 13  * 256 * 2);
    bf16_t* rbuf   = (bf16_t*)alloc((size_t)2 * 200 * 200 * 512 * 2);
    bf16_t* pooled = (bf16_t*)alloc((size_t)512 * 256  * 2);
    bf16_t* x1     = (bf16_t*)alloc((size_t)512 * 1024 * 2);
    bf16_t* x2     = (bf16_t*)alloc((size_t)512 * 1024 * 2);
    float*  logits = (float*)alloc((size_t)512 * 8 * 4);

    // ---- launch helpers ----
    auto pack = [&](const float* src, bf16_t* dst, int T, int K, int Nsrc, int Npad) {
        size_t total = (size_t)T * (Npad / 16) * (K / 32) * 512;
        pack_weights_kernel<<<(unsigned)((total + 255) / 256), 256, 0, stream>>>(
            src, dst, T, K, Nsrc, Npad);
    };
    auto gemm_grid = [](int M, int Npad, int JT) {
        return dim3((unsigned)(((M + 15) / 16 + 3) / 4), (unsigned)(Npad / (16 * JT)));
    };

    // ---- 1) pack all weights into WMMA fragment layout ----
    pack(lw2, pw_l2, 1, 256,  256, 256);
    pack(lw3, pw_l3, 1, 512,  256, 256);
    pack(lw4, pw_l4, 1, 1024, 256, 256);
    pack(lw5, pw_l5, 1, 2048, 256, 256);
    for (int l = 0; l < 4; ++l)
        pack(fw + (size_t)l * 9 * 256 * 256, pw_f[l], 9, 256, 256, 256);
    pack(p6w,  pw_p6,  9, 256, 256, 256);
    pack(rpnw, pw_rpn, 9, 256, 512, 512);
    pack(clsw, pw_cls, 1, 512, 3,  16);
    pack(regw, pw_reg, 1, 512, 12, 16);
    pack(fc1w, pw_fc1, 1, 256,  1024, 1024);
    pack(fc2w, pw_fc2, 1, 1024, 1024, 1024);
    pack(hclsw, pw_hcl, 1, 1024, 8,  16);
    pack(hregw, pw_hrg, 1, 1024, 32, 32);

    // ---- 2) lateral 1x1 convs: fp32 feats streamed once (JT=16 => grid.y=1) ----
    wmma_gemm_kernel<float, bf16_t, 16><<<gemm_grid(2*200*200, 256, 16), 128, 0, stream>>>(
        feat2, 256,  pw_l2, lb2, l2, 256, 2*200*200, 256,  256, 0);
    wmma_gemm_kernel<float, bf16_t, 16><<<gemm_grid(2*100*100, 256, 16), 128, 0, stream>>>(
        feat3, 512,  pw_l3, lb3, l3, 256, 2*100*100, 512,  256, 0);
    wmma_gemm_kernel<float, bf16_t, 16><<<gemm_grid(2*50*50, 256, 16), 128, 0, stream>>>(
        feat4, 1024, pw_l4, lb4, l4, 256, 2*50*50,   1024, 256, 0);
    wmma_gemm_kernel<float, bf16_t, 16><<<gemm_grid(2*25*25, 256, 16), 128, 0, stream>>>(
        feat5, 2048, pw_l5, lb5, l5, 256, 2*25*25,   2048, 256, 0);

    // ---- 3) top-down pathway: m = l + up2(m_upper), in place (bf16) ----
    {
        size_t n4 = (size_t)2 * 50 * 50 * 256;
        upadd_kernel<<<(unsigned)((n4 + 255) / 256), 256, 0, stream>>>(l4, l5, 2, 50, 50, 256);
        size_t n3 = (size_t)2 * 100 * 100 * 256;
        upadd_kernel<<<(unsigned)((n3 + 255) / 256), 256, 0, stream>>>(l3, l4, 2, 100, 100, 256);
        size_t n2 = (size_t)2 * 200 * 200 * 256;
        upadd_kernel<<<(unsigned)((n2 + 255) / 256), 256, 0, stream>>>(l2, l3, 2, 200, 200, 256);
    }

    // ---- 4) FPN 3x3 convs (implicit GEMM, async-LDS B staging) + p6 ----
    auto conv = [&](const bf16_t* X, int Nb, int H, int W,
                    const bf16_t* Wp, const float* bias, bf16_t* Y,
                    int Ho, int Wo, int Cout, int stride, int act) {
        int M = Nb * Ho * Wo;
        dim3 grid((unsigned)(((M + 15) / 16 + 3) / 4), (unsigned)(Cout / 256));
        wmma_conv3x3_kernel<16><<<grid, 128, 0, stream>>>(
            X, Nb, H, W, 256, Wp, bias, Y, Ho, Wo, Cout, stride, act);
    };
    conv(l2, 2, 200, 200, pw_f[0], fb + 0,   p2, 200, 200, 256, 1, 0);
    conv(l3, 2, 100, 100, pw_f[1], fb + 256, p3, 100, 100, 256, 1, 0);
    conv(l4, 2, 50,  50,  pw_f[2], fb + 512, p4, 50,  50,  256, 1, 0);
    conv(l5, 2, 25,  25,  pw_f[3], fb + 768, p5, 25,  25,  256, 1, 0);
    conv(p5, 2, 25,  25,  pw_p6,   p6b,      p6, 13,  13,  256, 2, 0);

    // ---- output offsets ----
    const int Hs[5] = {200, 100, 50, 25, 13};
    size_t clsOff[5], regOff[5], off = 0;
    for (int l = 0; l < 5; ++l) { clsOff[l] = off; off += (size_t)2 * Hs[l] * Hs[l] * 3; }
    for (int l = 0; l < 5; ++l) { regOff[l] = off; off += (size_t)2 * Hs[l] * Hs[l] * 12; }
    size_t roiClsOff = off; off += (size_t)512 * 8;
    size_t roiRegOff = off; off += (size_t)512 * 32;
    size_t anchorOff = off;

    // ---- 5) RPN per level: r = relu(conv3x3), cls = sigmoid(r@W), reg = r@W ----
    bf16_t* plist[5] = {p2, p3, p4, p5, p6};
    for (int l = 0; l < 5; ++l) {
        int H = Hs[l];
        conv(plist[l], 2, H, H, pw_rpn, rpnb, rbuf, H, H, 512, 1, 1);
        int M = 2 * H * H;
        wmma_gemm_kernel<bf16_t, float, 1><<<gemm_grid(M, 16, 1), 128, 0, stream>>>(
            rbuf, 512, pw_cls, clsb, out + clsOff[l], 3,  M, 512, 3,  2);
        wmma_gemm_kernel<bf16_t, float, 1><<<gemm_grid(M, 16, 1), 128, 0, stream>>>(
            rbuf, 512, pw_reg, regb, out + regOff[l], 12, M, 512, 12, 0);
    }

    // ---- 6) ROI align (crop_and_resize 7x7 + mean) on p4[0] ----
    roi_pool_kernel<<<512, 256, 0, stream>>>(p4, rois, pooled);

    // ---- 7) ROI head MLP + outputs ----
    wmma_gemm_kernel<bf16_t, bf16_t, 16><<<gemm_grid(512, 1024, 16), 128, 0, stream>>>(
        pooled, 256,  pw_fc1, fc1b, x1, 1024, 512, 256,  1024, 1);
    wmma_gemm_kernel<bf16_t, bf16_t, 16><<<gemm_grid(512, 1024, 16), 128, 0, stream>>>(
        x1,     1024, pw_fc2, fc2b, x2, 1024, 512, 1024, 1024, 1);
    wmma_gemm_kernel<bf16_t, float, 1><<<gemm_grid(512, 16, 1), 128, 0, stream>>>(
        x2,     1024, pw_hcl, hclsb, logits, 8, 512, 1024, 8, 0);
    softmax8_kernel<<<2, 256, 0, stream>>>(logits, out + roiClsOff);
    wmma_gemm_kernel<bf16_t, float, 2><<<gemm_grid(512, 32, 2), 128, 0, stream>>>(
        x2,     1024, pw_hrg, hregb, out + roiRegOff, 32, 512, 1024, 32, 0);

    // ---- 8) anchors ----
    anchors_kernel<<<(159882 + 255) / 256, 256, 0, stream>>>(out + anchorOff);
}
